// RemovalTimePredictor_74457553044346
// MI455X (gfx1250) — matrix operand order repaired and checked by reading, not compiled, not run on hardware
//
#include <hip/hip_runtime.h>

// ---------------------------------------------------------------------------
// EGAT (9-layer) forward for MI455X / gfx1250, wave32.
//  - GEMM core: v_wmma_f32_16x16x32_bf16, fp32 accumulate.
//  - Weight panels staged into LDS per block (8x reuse) via the Tensor Data
//    Mover (tensor_load_to_lds + s_wait_tensorcnt), cooperative-load fallback.
//  - Register double-buffering so WMMA issue overlaps operand fetch.
//
// Input flattening assumption (setup_inputs dict order; params per layer in
// insertion order Wn, bn, Wni, Wfij, Wnj, attn, bias):
//   d_in[0] node_f [10000,3] f32      d_in[1] edge_f [50000,1] f32
//   d_in[2] src    [50000]   i32      d_in[3] dst    [50000]   i32
//   d_in[4 + 7l + 0..6] = Wn, bn, Wni, Wfij, Wnj, attn, bias  (layer l)
// Output: d_out = concat(h_final [10000], f_final [50000]) f32.
// ---------------------------------------------------------------------------

typedef __bf16 bf16_t;
typedef __attribute__((ext_vector_type(16))) __bf16 v16bf;
typedef __attribute__((ext_vector_type(8)))  float  v8f;
typedef __attribute__((ext_vector_type(4)))  unsigned u32x4;
typedef __attribute__((ext_vector_type(8)))  unsigned u32x8;

#define N_NODES 10000
#define N_EDGES 50000
#define LEAKY   0.01f

#ifndef USE_TDM
#define USE_TDM 1        // flip to 0 if tensor_load_to_lds fails to assemble
#endif

// ------------------------------ utility kernels ----------------------------

__global__ void k_fill_zero(unsigned* __restrict__ p, long n) {
  long i = (long)blockIdx.x * blockDim.x + threadIdx.x;
  long st = (long)gridDim.x * blockDim.x;
  for (; i < n; i += st) p[i] = 0u;
}

__global__ void k_copy_f32(const float* __restrict__ a, float* __restrict__ o, long n) {
  long i = (long)blockIdx.x * blockDim.x + threadIdx.x;
  long st = (long)gridDim.x * blockDim.x;
  for (; i < n; i += st) o[i] = a[i];
}

// activations f32 -> bf16, zero-padded to Kpad, optional fused ReLU
__global__ void k_conv_a(const float* __restrict__ X, bf16_t* __restrict__ A,
                         int M, int Kin, int Kpad, int relu) {
  long n = (long)M * Kpad;
  long i = (long)blockIdx.x * blockDim.x + threadIdx.x;
  long st = (long)gridDim.x * blockDim.x;
  for (; i < n; i += st) {
    int  k   = (int)(i % Kpad);
    long row = i / Kpad;
    float v = 0.0f;
    if (k < Kin) v = X[row * Kin + k];
    if (relu && v < 0.0f) v = 0.0f;
    A[i] = (bf16_t)v;
  }
}

// weights f32 [Kin,Nc] -> transposed bf16 [Npad,Kpad]: Wt[n*Kpad+k] = W[k*Nc+n]
__global__ void k_conv_w(const float* __restrict__ W, bf16_t* __restrict__ Wt,
                         int Kin, int Nc, int Kpad, int Npad) {
  long n = (long)Npad * Kpad;
  long i = (long)blockIdx.x * blockDim.x + threadIdx.x;
  long st = (long)gridDim.x * blockDim.x;
  for (; i < n; i += st) {
    int col = (int)(i / Kpad);   // n index
    int k   = (int)(i % Kpad);
    float v = (col < Nc && k < Kin) ? W[(long)k * Nc + col] : 0.0f;
    Wt[i] = (bf16_t)v;
  }
}

// ------------------------------ WMMA GEMM ----------------------------------
// C[M,Nc] = A[M,Kpad](bf16) @ Wt^T (+bias); Wt stored [Npad,Kpad] bf16.
// Block: 256 thr = 8 waves; block tile = 128 rows x 64 cols.
// B panel staged in LDS per K-chunk (TDM), each wave runs 4 WMMAs per K-step
// with register double-buffered operands.
#define GEMM_NT 4
#define KC      320        // K-chunk elems: 64*320*2 = 40 KB LDS, 10 WMMA steps

#if USE_TDM
// Issue a 2D TDM load: tile [kcur x 64 rows] of bf16, row stride Kpad elems,
// global tile start gsrc -> LDS offset 0 (lds_base is the kernel's only static
// LDS object, so its offset within the workgroup allocation is 0).
// D# layout per CDNA5 ISA (2-group 2D form).
// IMPORTANT: lds_base is passed as an asm operand so the LDS array's address
// escapes -- otherwise the compiler proves "no store to Bs" and folds every
// LDS read of the panel to poison (observed in round 2: ds=0, shared B reg).
__device__ __forceinline__ void tdm_load_panel(const bf16_t* gsrc, bf16_t* lds_base,
                                               int Kpad, int Npad, int kcur) {
  unsigned long long ga = (unsigned long long)gsrc;
  u32x4 g0;
  g0[0] = 1u;                                   // count=1, user-mode, no gather
  g0[1] = 0u;                                   // lds_addr = 0 (Bs at LDS base)
  g0[2] = (unsigned)ga;                         // global_addr[31:0]
  g0[3] = (unsigned)((ga >> 32) & 0x01FFFFFFull) | (2u << 30);  // [56:32]|type=2
  u32x8 g1;
  g1[0] = 1u << 16;                             // data_size=1 -> 2-byte elems
  g1[1] = ((unsigned)Kpad & 0xFFFFu) << 16;     // tensor_dim0[15:0]
  g1[2] = (((unsigned)Kpad >> 16) & 0xFFFFu) |  // tensor_dim0[31:16]
          (((unsigned)Npad & 0xFFFFu) << 16);   // tensor_dim1[15:0]
  g1[3] = (((unsigned)Npad >> 16) & 0xFFFFu) |  // tensor_dim1[31:16]
          (((unsigned)kcur & 0xFFFFu) << 16);   // tile_dim0 = kcur
  g1[4] = 64u;                                  // tile_dim1 = 64, tile_dim2 = 0
  g1[5] = (unsigned)Kpad;                       // tensor_dim0_stride[31:0]
  g1[6] = 0u;                                   // stride hi, dim1_stride lo16
  g1[7] = 0u;
  asm volatile("tensor_load_to_lds %0, %1"
               :: "s"(g0), "s"(g1), "s"(lds_base)   // %2 unused: escapes LDS ptr
               : "memory");
}
#endif

__global__ void k_gemm_wmma(const bf16_t* __restrict__ A,
                            const bf16_t* __restrict__ Bt,
                            const float*  __restrict__ bias,
                            float* __restrict__ C,
                            int M, int Kpad, int Npad, int Nc) {
  __shared__ bf16_t Bs[64 * KC];

  const int lane  = threadIdx.x & 31;
  const int wave  = threadIdx.x >> 5;
  const int nbase = blockIdx.y * 64;
  const int row0  = (blockIdx.x * 8 + wave) * 16;
  const int half  = lane >> 4;                 // 0 | 1
  const int l16   = lane & 15;
  const bool live = (row0 < M);                // wave-uniform

  // 16-bit A 16x32 layout: lane holds row l16; K = (half?16:0)+0..15.
  const bf16_t* arow = A + (long)(row0 + l16) * Kpad + half * 16;

  v8f acc[GEMM_NT] = {};

  for (int kc = 0; kc < Kpad; kc += KC) {
    const int kcur = (Kpad - kc < KC) ? (Kpad - kc) : KC;   // multiple of 32
    __syncthreads();                           // LDS reuse across chunks
#if USE_TDM
    if (threadIdx.x == 0)
      tdm_load_panel(Bt + (long)nbase * Kpad + kc, Bs, Kpad, Npad, kcur);
    __builtin_amdgcn_s_wait_tensorcnt(0);      // wave0 waits; others trivially 0
#else
    for (int r = wave; r < 64; r += 8)
      for (int k8 = lane * 8; k8 < kcur; k8 += 32 * 8)
        *(uint4*)(Bs + (long)r * kcur + k8) =
            *(const uint4*)(Bt + (long)(nbase + r) * Kpad + kc + k8);
#endif
    __syncthreads();

    if (live) {
      if (kc + KC < Kpad) __builtin_prefetch(arow + kc + KC, 0, 1);
      // B 32x16 frag, same contiguous k order as A: lane holds col l16; K = (half?16:0)+0..15,
      // contiguous in LDS because panel rows are K-major.
      const bf16_t* bcol[GEMM_NT];
#pragma unroll
      for (int t = 0; t < GEMM_NT; ++t)
        bcol[t] = Bs + (long)(16 * t + l16) * kcur + half * 16;
      const bf16_t* ak = arow + kc;

      union AF { v16bf v; uint4 q[2]; };
      AF a_cur, a_nxt;
      v16bf b_cur[GEMM_NT], b_nxt[GEMM_NT];
      a_cur.q[0] = *(const uint4*)(ak);
      a_cur.q[1] = *(const uint4*)(ak + 8);
#pragma unroll
      for (int t = 0; t < GEMM_NT; ++t) b_cur[t] = *(const v16bf*)(bcol[t]);

      for (int kb = 0; kb < kcur; kb += 32) {
        const int kn = kb + 32;
        if (kn < kcur) {                       // issue next frags before WMMAs
          a_nxt.q[0] = *(const uint4*)(ak + kn);
          a_nxt.q[1] = *(const uint4*)(ak + kn + 8);
#pragma unroll
          for (int t = 0; t < GEMM_NT; ++t)
            b_nxt[t] = *(const v16bf*)(bcol[t] + kn);
        }
#pragma unroll
        for (int t = 0; t < GEMM_NT; ++t)
          acc[t] = __builtin_amdgcn_wmma_f32_16x16x32_bf16(
              false, a_cur.v, false, b_cur[t], (short)0, acc[t], false, false);
        a_cur = a_nxt;
#pragma unroll
        for (int t = 0; t < GEMM_NT; ++t) b_cur[t] = b_nxt[t];
      }
    }
  }

  if (live) {
    // C/D layout: VGPR r, lane -> (row0 + r + half*8, nbase + 16t + l16)
#pragma unroll
    for (int t = 0; t < GEMM_NT; ++t) {
      int col = nbase + 16 * t + l16;
      if (col < Nc) {
        float badd = bias ? bias[col] : 0.0f;
#pragma unroll
        for (int r = 0; r < 8; ++r) {
          int row = row0 + r + half * 8;
          C[(long)row * Nc + col] = acc[t][r] + badd;
        }
      }
    }
  }
}

// ------------------------- edge features + logits --------------------------
__global__ void k_edge(const float* __restrict__ f_ni, const float* __restrict__ f_nj,
                       const float* __restrict__ f_fij,
                       const float* __restrict__ bias, const float* __restrict__ attn,
                       const int* __restrict__ src, const int* __restrict__ dst,
                       float* __restrict__ fe, float* __restrict__ logits,
                       int E, int h, int fo) {
  int e = blockIdx.x, t = threadIdx.x;
  if (e >= E || t >= h) return;
  int  F  = h * fo;
  long sb = (long)src[e] * F, db = (long)dst[e] * F, eb = (long)e * F;
  float acc = 0.0f;
  for (int j = 0; j < fo; ++j) {
    int idx = t * fo + j;
    float v = f_ni[sb + idx] + f_nj[db + idx] + f_fij[eb + idx] + bias[idx];
    v = v > 0.0f ? v : LEAKY * v;     // LeakyReLU(0.01)
    fe[eb + idx] = v;
    acc += v * attn[idx];             // attn[t][j]
  }
  logits[(long)e * h + t] = acc;
}

// ---------------------- edge softmax over destination ----------------------
__device__ inline unsigned enc_f32(float f) {
  unsigned u = __float_as_uint(f);
  return (u & 0x80000000u) ? ~u : (u | 0x80000000u);
}
__device__ inline float dec_f32(unsigned k) {
  unsigned u = (k & 0x80000000u) ? (k & 0x7FFFFFFFu) : ~k;
  return __uint_as_float(u);
}

__global__ void k_segmax(const float* __restrict__ logits, const int* __restrict__ dst,
                         unsigned* __restrict__ menc, int E, int h) {
  long n = (long)E * h;
  long i = (long)blockIdx.x * blockDim.x + threadIdx.x;
  long st = (long)gridDim.x * blockDim.x;
  for (; i < n; i += st) {
    int e = (int)(i / h), t = (int)(i % h);
    atomicMax(&menc[(long)dst[e] * h + t], enc_f32(logits[i]));
  }
}

__global__ void k_segexp(float* __restrict__ logits, const int* __restrict__ dst,
                         const unsigned* __restrict__ menc, float* __restrict__ s,
                         int E, int h) {
  long n = (long)E * h;
  long i = (long)blockIdx.x * blockDim.x + threadIdx.x;
  long st = (long)gridDim.x * blockDim.x;
  for (; i < n; i += st) {
    int e = (int)(i / h), t = (int)(i % h);
    float m  = dec_f32(menc[(long)dst[e] * h + t]);
    float ex = __expf(logits[i] - m);
    logits[i] = ex;                                  // reuse buffer for exp()
    atomicAdd(&s[(long)dst[e] * h + t], ex);
  }
}

// ------------------- weighted message scatter-aggregation ------------------
__global__ void k_aggregate(const float* __restrict__ h_src, const float* __restrict__ ex,
                            const float* __restrict__ s,
                            const int* __restrict__ src, const int* __restrict__ dst,
                            float* __restrict__ h_out, int E, int h, int dno) {
  int e = blockIdx.x;
  if (e >= E) return;
  int F = h * dno;
  int se = src[e], de = dst[e];
  for (int idx = threadIdx.x; idx < F; idx += blockDim.x) {
    int head = idx / dno;
    float a = ex[(long)e * h + head] / s[(long)de * h + head];
    atomicAdd(&h_out[(long)de * F + idx], h_src[(long)se * F + idx] * a);
  }
}

// ------------------------------- host side ---------------------------------

static inline int nblk(long n, int t) {
  long b = (n + t - 1) / t;
  return (int)(b > 1048576 ? 1048576 : (b < 1 ? 1 : b));
}

extern "C" void kernel_launch(void* const* d_in, const int* in_sizes, int n_in,
                              void* d_out, int out_size, void* d_ws, size_t ws_size,
                              hipStream_t stream) {
  (void)in_sizes; (void)n_in; (void)out_size; (void)ws_size;
  const float* node_f = (const float*)d_in[0];
  const float* edge_f = (const float*)d_in[1];
  const int*   src    = (const int*)d_in[2];
  const int*   dst    = (const int*)d_in[3];
  const int N = N_NODES, E = N_EDGES;

  static const int S_NI[9] = {3,900,900,900,900,900,900,900,900};
  static const int S_NE[9] = {1,900,900,900,900,900,900,900,900};
  static const int S_DO[9] = {15,15,15,15,15,15,15,15,1};
  static const int S_FO[9] = {15,15,15,15,15,15,15,15,1};
  static const int S_H [9] = {60,60,60,60,60,60,60,60,1};

  char* ws = (char*)d_ws;
  size_t off = 0;
  auto walloc = [&](size_t bytes) -> void* {
    void* p = ws + off;
    off += (bytes + 255) & ~(size_t)255;
    return p;
  };
  const int KPAD_MAX = 928;   // ceil(900/32)*32
  const int NPAD_MAX = 960;   // ceil(900/64)*64
  bf16_t* A_node = (bf16_t*)walloc((size_t)N * KPAD_MAX * 2);
  bf16_t* A_edge = (bf16_t*)walloc((size_t)E * KPAD_MAX * 2);
  bf16_t* Wt_n   = (bf16_t*)walloc((size_t)NPAD_MAX * KPAD_MAX * 2);
  bf16_t* Wt_ni  = (bf16_t*)walloc((size_t)NPAD_MAX * KPAD_MAX * 2);
  bf16_t* Wt_nj  = (bf16_t*)walloc((size_t)NPAD_MAX * KPAD_MAX * 2);
  bf16_t* Wt_fij = (bf16_t*)walloc((size_t)NPAD_MAX * KPAD_MAX * 2);
  float* f_ni   = (float*)walloc((size_t)N * 900 * 4);
  float* f_nj   = (float*)walloc((size_t)N * 900 * 4);
  float* h_srcb = (float*)walloc((size_t)N * 900 * 4);
  float* f_fij  = (float*)walloc((size_t)E * 900 * 4);
  float* fe     = (float*)walloc((size_t)E * 900 * 4);   // edge feats / f_final
  float* logits = (float*)walloc((size_t)E * 60 * 4);
  unsigned* menc = (unsigned*)walloc((size_t)N * 60 * 4);
  float* sden   = (float*)walloc((size_t)N * 60 * 4);
  float* h_out  = (float*)walloc((size_t)N * 900 * 4);   // node feats / h_final

  for (int l = 0; l < 9; ++l) {
    const int ni = S_NI[l], ne = S_NE[l], dno = S_DO[l], fo = S_FO[l], h = S_H[l];
    const int F  = h * fo;    // edge-feature / attention width
    const int Fn = h * dno;   // node-message width
    const int KpadN = ((ni + 31) / 32) * 32;
    const int KpadE = ((ne + 31) / 32) * 32;
    const int NpadF  = ((F  + 63) / 64) * 64;
    const int NpadFn = ((Fn + 63) / 64) * 64;

    const float* Wn   = (const float*)d_in[4 + l * 7 + 0];
    const float* bn   = (const float*)d_in[4 + l * 7 + 1];
    const float* Wni  = (const float*)d_in[4 + l * 7 + 2];
    const float* Wfij = (const float*)d_in[4 + l * 7 + 3];
    const float* Wnj  = (const float*)d_in[4 + l * 7 + 4];
    const float* attn = (const float*)d_in[4 + l * 7 + 5];
    const float* bias = (const float*)d_in[4 + l * 7 + 6];

    // weight conversion (transpose + pad + bf16)
    k_conv_w<<<nblk((long)NpadF * KpadN, 256), 256, 0, stream>>>(Wni,  Wt_ni,  ni, F,  KpadN, NpadF);
    k_conv_w<<<nblk((long)NpadF * KpadN, 256), 256, 0, stream>>>(Wnj,  Wt_nj,  ni, F,  KpadN, NpadF);
    k_conv_w<<<nblk((long)NpadF * KpadE, 256), 256, 0, stream>>>(Wfij, Wt_fij, ne, F,  KpadE, NpadF);
    k_conv_w<<<nblk((long)NpadFn * KpadN, 256), 256, 0, stream>>>(Wn,  Wt_n,   ni, Fn, KpadN, NpadFn);

    // activation conversion (ReLU fused for layers >= 1)
    const float* nf_src = (l == 0) ? node_f : h_out;
    const float* ef_src = (l == 0) ? edge_f : fe;
    const int nf_k = (l == 0) ? 3 : 900;
    const int ef_k = (l == 0) ? 1 : 900;
    k_conv_a<<<nblk((long)N * KpadN, 256), 256, 0, stream>>>(nf_src, A_node, N, nf_k, KpadN, l != 0);
    k_conv_a<<<nblk((long)E * KpadE, 256), 256, 0, stream>>>(ef_src, A_edge, E, ef_k, KpadE, l != 0);

    // WMMA GEMMs
    dim3 blk(256);
    dim3 gF ((N / 16 + 7) / 8, NpadF  / 64);
    dim3 gFn((N / 16 + 7) / 8, NpadFn / 64);
    dim3 gE ((E / 16 + 7) / 8, NpadF  / 64);
    k_gemm_wmma<<<gF,  blk, 0, stream>>>(A_node, Wt_ni,  nullptr, f_ni,   N, KpadN, NpadF,  F);
    k_gemm_wmma<<<gF,  blk, 0, stream>>>(A_node, Wt_nj,  nullptr, f_nj,   N, KpadN, NpadF,  F);
    k_gemm_wmma<<<gFn, blk, 0, stream>>>(A_node, Wt_n,   bn,      h_srcb, N, KpadN, NpadFn, Fn);
    k_gemm_wmma<<<gE,  blk, 0, stream>>>(A_edge, Wt_fij, nullptr, f_fij,  E, KpadE, NpadF,  F);

    // edge features + attention logits
    k_edge<<<E, 64, 0, stream>>>(f_ni, f_nj, f_fij, bias, attn, src, dst,
                                 fe, logits, E, h, fo);

    // edge softmax over dst + weighted scatter aggregation
    k_fill_zero<<<nblk((long)N * h, 256), 256, 0, stream>>>(menc, (long)N * h);
    k_fill_zero<<<nblk((long)N * h, 256), 256, 0, stream>>>((unsigned*)sden, (long)N * h);
    k_fill_zero<<<nblk((long)N * Fn, 256), 256, 0, stream>>>((unsigned*)h_out, (long)N * Fn);
    k_segmax<<<nblk((long)E * h, 256), 256, 0, stream>>>(logits, dst, menc, E, h);
    k_segexp<<<nblk((long)E * h, 256), 256, 0, stream>>>(logits, dst, menc, sden, E, h);
    k_aggregate<<<E, 128, 0, stream>>>(h_srcb, logits, sden, src, dst, h_out, E, h, dno);
  }

  // outputs: h_final (raw, [N]) then f_final (post-LeakyReLU edge feats, [E])
  k_copy_f32<<<nblk(N, 256), 256, 0, stream>>>(h_out, (float*)d_out, N);
  k_copy_f32<<<nblk(E, 256), 256, 0, stream>>>(fe, (float*)d_out + N, E);
}
